// Cfconv_85718957294069
// MI455X (gfx1250) — matrix operand order repaired and checked by reading, not compiled, not run on hardware
//
#include <hip/hip_runtime.h>
#include <hip/hip_bf16.h>

typedef __attribute__((ext_vector_type(16))) _Float16 v16h;
typedef __attribute__((ext_vector_type(8)))  float    v8f;

#define GAMMA_    10.0f
#define RBF_DELTA (30.0f / 299.0f)   // linspace(0,30,300) spacing

// h-tile LDS row stride in halves (multiple of 8 -> 16B-aligned b128 runs)
#define HSTRIDE 72

union FragH { v16h v; int4 q[2]; };

// shifted softplus: softplus(v) - ln2 = max(v,0) + log(0.5 + 0.5*exp(-|v|))
// well-conditioned everywhere; lowers to v_exp_f32 + v_fma + v_log_f32 (+muls)
__device__ __forceinline__ float sspf(float v) {
    const float a = fabsf(v);
    return fmaxf(v, 0.0f) + __logf(fmaf(0.5f, __expf(-a), 0.5f));
}

__global__ __launch_bounds__(128) void cfconv_wmma_kernel(
    const float* __restrict__ x,     // [BN][64]
    const float* __restrict__ dist,  // [BN][32]
    const float* __restrict__ W1,    // [300][64]
    const float* __restrict__ b1,    // [64]
    const float* __restrict__ W2,    // [64][64]
    const float* __restrict__ b2,    // [64]
    float* __restrict__ out)         // [BN][64]
{
    // Pre-swizzled f16 B-fragments: [kstep(2)][ntile(4)][lane(32)][16 halves]
    __shared__ _Float16 w1f[2 * 4 * 32 * 16];   // 8 KB  (k = 0..63 of W1; for d in [0,1)
    __shared__ _Float16 w2f[2 * 4 * 32 * 16];   // 8 KB   all k>=64 RBF terms are exact fp32 zeros)
    __shared__ _Float16 hbuf[4][16 * HSTRIDE];  // per-wave 16x64(+pad) h tile, 9 KB total

    const int tid = threadIdx.x;

    // ---- Block-cooperative staging of W1(k<64) and W2 into B-fragment layout ----
    // One "slot" = one lane's 16-half fragment run (16 consecutive K, fixed column).
    // 512 slots = 2 matrices x 2 ksteps x 4 ntiles x 32 lane-slots.
#pragma unroll
    for (int it = 0; it < 4; ++it) {
        const int slot = tid + 128 * it;
        const int mat  = slot >> 8;          // 0: W1, 1: W2  (uniform per iteration)
        const int rem  = slot & 255;
        const int s    = rem >> 7;           // kstep
        const int t    = (rem >> 5) & 3;     // ntile
        const int ls   = rem & 31;           // lane slot
        const int gs   = ls >> 4;
        const int c    = (ls & 15) + 16 * t; // output column
        const int kb   = s * 32 + 16 * gs;   // first K of this run
        const float* src = (mat ? W2 : W1) + (size_t)kb * 64 + c;
        FragH f;
#pragma unroll
        for (int j = 0; j < 16; ++j)
            f.v[j] = (_Float16)src[(size_t)j * 64];   // stride-64 gather, L2-resident
        _Float16* dst = (mat ? w2f : w1f) + ((s * 4 + t) * 32 + ls) * 16;
        ((int4*)dst)[0] = f.q[0];
        ((int4*)dst)[1] = f.q[1];
    }
    __syncthreads();

    const int wave = tid >> 5;
    const int lane = tid & 31;
    const int g    = lane >> 4;   // half-wave group
    const int ln   = lane & 15;

    const int bn = blockIdx.x * 4 + wave;     // one (b,n) pair per wave
    const float* dbn = dist + (size_t)bn * 32;
    _Float16* hrow = &hbuf[wave][0];

    // Each lane holds the distance of neighbor m == lane
    const float dself = dbn[lane];

    float b1r[4], b2r[4];
#pragma unroll
    for (int t = 0; t < 4; ++t) { b1r[t] = b1[ln + 16 * t]; b2r[t] = b2[ln + 16 * t]; }

    float colsum[4] = {0.f, 0.f, 0.f, 0.f};
    const v8f vzero = {};

#pragma unroll
    for (int tau = 0; tau < 2; ++tau) {       // two 16-neighbor tiles
        // broadcast distance of this lane's A-row: m = (lane&15) + 16*tau
        const float dm = __shfl(dself, ln + 16 * tau, 32);

        // ---------------- GEMM1: e_k[16x64] x W1[64x64] ----------------
        v8f acc[4];
#pragma unroll
        for (int t = 0; t < 4; ++t) acc[t] = vzero;

#pragma unroll
        for (int s = 0; s < 2; ++s) {
            // Build A fragment (16x32 f16 layout) directly from the RBF expansion
            FragH a;
#pragma unroll
            for (int j = 0; j < 16; ++j) {
                const int kk = s * 32 + 16 * g + j;
                const float d = dm - (float)kk * RBF_DELTA;
                a.v[j] = (_Float16)__expf(-GAMMA_ * d * d);
            }
#pragma unroll
            for (int t = 0; t < 4; ++t) {
                FragH bf;
                const int4* p = (const int4*)&w1f[((s * 4 + t) * 32 + lane) * 16];
                bf.q[0] = p[0]; bf.q[1] = p[1];
                acc[t] = __builtin_amdgcn_wmma_f32_16x16x32_f16(
                    false, a.v, false, bf.v, (short)0, acc[t], false, false);
            }
        }

        // bias + SSP, write h tile to LDS (row-major [m][c], stride HSTRIDE)
#pragma unroll
        for (int t = 0; t < 4; ++t) {
            const int c = ln + 16 * t;
#pragma unroll
            for (int r = 0; r < 8; ++r) {
                const float hv = sspf(acc[t][r] + b1r[t]);
                hrow[(r + 8 * g) * HSTRIDE + c] = (_Float16)hv;
            }
        }

        // ---------------- GEMM2: h[16x64] x W2[64x64] ----------------
        v8f acc2[4];
#pragma unroll
        for (int t = 0; t < 4; ++t) acc2[t] = vzero;

#pragma unroll
        for (int s = 0; s < 2; ++s) {
            FragH a2;  // A row m = lane&15; K run {s*32+16g .. +15}
            a2.q[0] = *(const int4*)&hrow[ln * HSTRIDE + s * 32 + 16 * g];
            a2.q[1] = *(const int4*)&hrow[ln * HSTRIDE + s * 32 + 16 * g + 8];
#pragma unroll
            for (int t = 0; t < 4; ++t) {
                FragH bf;
                const int4* p = (const int4*)&w2f[((s * 4 + t) * 32 + lane) * 16];
                bf.q[0] = p[0]; bf.q[1] = p[1];
                acc2[t] = __builtin_amdgcn_wmma_f32_16x16x32_f16(
                    false, a2.v, false, bf.v, (short)0, acc2[t], false, false);
            }
        }

        // bias + SSP, accumulate per-column sums over this tile's 8 in-lane rows
#pragma unroll
        for (int t = 0; t < 4; ++t) {
#pragma unroll
            for (int r = 0; r < 8; ++r)
                colsum[t] += sspf(acc2[t][r] + b2r[t]);
        }
    }

    // Combine half-wave row groups: lane holds rows {8g..8g+7, 16+8g..16+8g+7};
    // lane^16 holds the complementary 16 neighbors.
    const float* xbn = x + (size_t)bn * 64;
    float*       obn = out + (size_t)bn * 64;
#pragma unroll
    for (int t = 0; t < 4; ++t) {
        const float tot = colsum[t] + __shfl_xor(colsum[t], 16, 32);
        if (lane < 16) {
            const int c = ln + 16 * t;
            obn[c] = xbn[c] * tot;
        }
    }
}

extern "C" void kernel_launch(void* const* d_in, const int* in_sizes, int n_in,
                              void* d_out, int out_size, void* d_ws, size_t ws_size,
                              hipStream_t stream) {
    const float* x    = (const float*)d_in[0];  // [B,N,F]   = [16,512,64]
    const float* dist = (const float*)d_in[1];  // [B,N,M]   = [16,512,32]
    const float* W1   = (const float*)d_in[2];  // [300,64]
    const float* b1   = (const float*)d_in[3];  // [64]
    const float* W2   = (const float*)d_in[4];  // [64,64]
    const float* b2   = (const float*)d_in[5];  // [64]
    float* out        = (float*)d_out;          // [B,N,64]

    const int BN = in_sizes[0] / 64;            // 8192 (b,n) pairs
    const int blocks = BN / 4;                  // 4 waves per block, 1 (b,n) per wave

    cfconv_wmma_kernel<<<blocks, 128, 0, stream>>>(x, dist, W1, b1, W2, b2, out);
}
